// DistributionTokenizer_1228360646753
// MI455X (gfx1250) — compile-verified, matrix operand order checked by reading, not run on hardware
//
#include <hip/hip_runtime.h>
#include <stdint.h>

#define NBINS   64
#define FDIM    1024
#define THREADS 256
#define NWAVES  (THREADS / 32)   // wave32 on gfx1250

typedef __attribute__((ext_vector_type(16))) _Float16 v16h;
typedef __attribute__((ext_vector_type(8)))  float    v8f;

// One workgroup per (B,T) row.
//  1. Async-stage the 1024-float row into LDS (global_load_async_to_lds_b128).
//  2. Bucketize: arithmetic guess (boundaries are a uniform linspace) plus a
//     BRANCHLESS +/-1 correction against the true boundary values in LDS ->
//     exact searchsorted(side='right'), 2 LDS probes per element, no loops.
//  3. Per-wave LDS sub-histograms (atomic contention stays intra-wave).
//  4. Cross-wave reduction + 1/1024 normalization on the matrix pipe:
//     v_wmma_f32_16x16x32_f16 is exact here (counts <= 128, scale = 2^-10).
__global__ __launch_bounds__(THREADS)
void dist_tokenizer_hist(const float* __restrict__ x,
                         const float* __restrict__ bnd,
                         float* __restrict__ out,
                         int nbnd)
{
    __shared__ float    stage[FDIM];            // async-staged input row (4 KiB)
    __shared__ float    p[NBINS + 2];           // -inf, bnd[0..62], +inf
    __shared__ unsigned shist[NWAVES][NBINS];   // per-wave sub-histograms

    const int tid  = threadIdx.x;
    const int lane = tid & 31;
    const int wave = tid >> 5;
    const long long row = blockIdx.x;

    // ---- 1. issue async global->LDS copy of this row (16B per lane) ----
    {
        uint64_t gaddr = (uint64_t)(uintptr_t)(x + row * FDIM + tid * 4);
        uint32_t laddr = (uint32_t)(uintptr_t)(&stage[tid * 4]);
        asm volatile("global_load_async_to_lds_b128 %0, %1, off"
                     :: "v"(laddr), "v"(gaddr)
                     : "memory");
    }

    // ---- init LDS tables while the async copy is in flight ----
    if (tid == 0)              p[0] = -__builtin_inff();
    else if (tid <= nbnd)      p[tid] = bnd[tid - 1];
    else if (tid == nbnd + 1)  p[tid] = __builtin_inff();
    unsigned* shf = &shist[0][0];               // 512 entries, 2 per thread
    shf[tid]           = 0u;
    shf[tid + THREADS] = 0u;

    // uniform-grid parameters from the actual boundary array (scalar loads).
    // The guess they seed is off by at most +/-1 (linspace points are within
    // ulps of the ideal grid, bin width >> rounding error); the branchless
    // probe below makes the result exact w.r.t. the true float boundaries.
    const float b0      = bnd[0];
    const float invStep = (float)(nbnd - 1) / (bnd[nbnd - 1] - b0);

#if __has_builtin(__builtin_amdgcn_s_wait_asynccnt)
    __builtin_amdgcn_s_wait_asynccnt(0);
#else
    asm volatile("s_wait_asynccnt 0x0" ::: "memory");
#endif
    __syncthreads();

    // ---- 2+3. bucketize + per-wave histogram (4 elements per thread) ----
    const float4 v4 = *reinterpret_cast<const float4*>(&stage[tid * 4]);
    const float elems[4] = { v4.x, v4.y, v4.z, v4.w };
#pragma unroll
    for (int k = 0; k < 4; ++k) {
        const float v = elems[k];
        // guess for bin = #(bnd <= v); correct bin b has p[b] <= v < p[b+1]
        int b = (int)floorf((v - b0) * invStep) + 1;
        b = b < 0 ? 0 : (b > nbnd ? nbnd : b);
        // branchless +/-1 correction: the two conditions are mutually
        // exclusive (hi <= v implies lo <= v, and lo > v implies hi > v)
        const float lo = p[b];
        const float hi = p[b + 1];
        b += (hi <= v) ? 1 : 0;
        b -= (lo >  v) ? 1 : 0;
        atomicAdd(&shist[wave][b], 1u);         // ds_add_u32, intra-wave only
    }
    __syncthreads();

    // ---- 4. cross-wave reduction + 1/1024 normalization via WMMA ----
    // Waves 0..3 each own one 16-bin group. A = const 1/1024, so the result is
    // (1/1024) * sum_k B[k][n] independent of B's K-layout; B rows 0..7 carry
    // the 8 sub-histogram counts for this bin group.
    if (wave < 4) {
        const _Float16 w = (_Float16)(1.0f / (float)FDIM); // 2^-10, exact
        v16h a;
#pragma unroll
        for (int i = 0; i < 16; ++i) a[i] = w;

        v16h bm;
#pragma unroll
        for (int i = 0; i < 16; ++i) bm[i] = (_Float16)0.0f;
        if (lane < 16) {
#pragma unroll
            for (int j = 0; j < NWAVES; ++j)
                bm[j] = (_Float16)(float)shist[j][wave * 16 + lane];
        }

        v8f c = {};
        // 8 args: (neg_a, A, neg_b, B, c_mod, C, reuse_a, reuse_b)
        c = __builtin_amdgcn_wmma_f32_16x16x32_f16(
                false, a, false, bm, (short)0, c, false, false);

        // C/D layout: VGPR0, lanes 0..15 -> M=0, N=lane (all M rows identical)
        if (lane < 16)
            out[row * NBINS + wave * 16 + lane] = c[0];
    }
}

extern "C" void kernel_launch(void* const* d_in, const int* in_sizes, int n_in,
                              void* d_out, int out_size, void* d_ws, size_t ws_size,
                              hipStream_t stream) {
    (void)n_in; (void)out_size; (void)d_ws; (void)ws_size;
    const float* x   = (const float*)d_in[0];
    const float* bnd = (const float*)d_in[1];
    const int nbnd   = in_sizes[1];          // 63 boundaries
    const int rows   = in_sizes[0] / FDIM;   // B*T = 16384
    dist_tokenizer_hist<<<rows, THREADS, 0, stream>>>(x, bnd, (float*)d_out, nbnd);
}
